// GIN_84456236908864
// MI455X (gfx1250) — matrix operand order repaired and checked by reading, not compiled, not run on hardware
//
#include <hip/hip_runtime.h>

#define GIN_L 3
#define GIN_H 64
#define GIN_C 10
#define GIN_N 100000
#define GIN_E 1600000
#define GIN_G 512
#define WT_PITCH 72   // padded halves per row: 144B rows -> 16B aligned b128, conflict-free

typedef __attribute__((ext_vector_type(16))) _Float16 v16h;
typedef __attribute__((ext_vector_type(8)))  _Float16 v8h;
typedef __attribute__((ext_vector_type(8)))  float    v8f;

#ifndef __has_builtin
#define __has_builtin(x) 0
#endif
#if __has_builtin(__builtin_amdgcn_global_load_async_to_lds_b128) && \
    __has_builtin(__builtin_amdgcn_s_wait_asynccnt)
#define GIN_ASYNC_LDS 1
typedef __attribute__((ext_vector_type(4))) int v4i;
typedef __attribute__((address_space(1))) v4i gv4i;   // global (AS1) int4
typedef __attribute__((address_space(3))) v4i lv4i;   // LDS (AS3) int4
#endif

// ---------------------------------------------------------------- zero
__global__ __launch_bounds__(256) void gin_zero(float* __restrict__ p, int n) {
  int i = blockIdx.x * 256 + threadIdx.x;
  if (i < n) p[i] = 0.0f;
}

// ------------------------------------------------- edge aggregation (scatter-add)
__global__ __launch_bounds__(256) void gin_aggregate(
    const float* __restrict__ h, const int* __restrict__ src, const int* __restrict__ dst,
    const float* __restrict__ ew, float* __restrict__ agg) {
  unsigned tid = blockIdx.x * 256u + threadIdx.x;
  if (tid >= (unsigned)GIN_E * 16u) return;
  unsigned e  = tid >> 4;
  unsigned c4 = (tid & 15u) << 2;
  int s = src[e], d = dst[e];
  float w = ew[e];
  const float4 hv = *(const float4*)(h + (size_t)s * GIN_H + c4);
  float* p = agg + (size_t)d * GIN_H + c4;
  atomicAdd(p + 0, hv.x * w);
  atomicAdd(p + 1, hv.y * w);
  atomicAdd(p + 2, hv.z * w);
  atomicAdd(p + 3, hv.w * w);
}

// ------------------------------------------------- WMMA GEMM (N x 64) @ (64 x 64) + bias
// MODE 0: in = (1+eps)*h + agg       (GIN combine), then @W1+b1
// MODE 1: in = relu(t*scale + shift) (fused BN1+ReLU), then @W2+b2
// Epilogue: store f32 + per-column sum / sum-of-squares for the next BN.
template <int MODE>
__global__ __launch_bounds__(256) void gin_gemm(
    const float* __restrict__ in, const float* __restrict__ agg,
    const float* __restrict__ epsPtr,
    const float* __restrict__ scale, const float* __restrict__ shift,
    const float* __restrict__ W, const float* __restrict__ bias,
    float* __restrict__ out,
    float* __restrict__ colsum, float* __restrict__ colsq) {
  __shared__ _Float16 Wt[GIN_H * WT_PITCH];      // transposed: Wt[n*72 + k], f16
  __shared__ float sSc[GIN_H], sSh[GIN_H];

#if defined(GIN_ASYNC_LDS)
  // CDNA5 async global->LDS staging of the 16KB f32 weight block (ASYNCcnt path),
  // then transpose+convert to f16 in LDS.
  __shared__ float Wstage[GIN_H * GIN_H];
#pragma unroll
  for (int it = 0; it < 4; ++it) {
    int idx = (it * 256 + threadIdx.x) * 4;      // one b128 per lane per iter
    __builtin_amdgcn_global_load_async_to_lds_b128(
        (gv4i*)(W + idx), (lv4i*)(Wstage + idx), 0, 0);
  }
  __builtin_amdgcn_s_wait_asynccnt(0);
  __syncthreads();
  for (int idx = threadIdx.x; idx < GIN_H * GIN_H; idx += 256) {
    int k = idx >> 6, n = idx & 63;
    Wt[n * WT_PITCH + k] = (_Float16)Wstage[idx];
  }
#else
  for (int idx = threadIdx.x; idx < GIN_H * GIN_H; idx += 256) {
    int k = idx >> 6, n = idx & 63;
    Wt[n * WT_PITCH + k] = (_Float16)W[idx];
  }
#endif
  if (MODE == 1) {
    if (threadIdx.x < GIN_H) { sSc[threadIdx.x] = scale[threadIdx.x]; sSh[threadIdx.x] = shift[threadIdx.x]; }
  }
  __syncthreads();

  const int lane = threadIdx.x & 31;
  const int wv   = threadIdx.x >> 5;
  const int rowBase = (blockIdx.x * 8 + wv) * 16;
  if (rowBase >= GIN_N) return;   // strip-granular tail: EXEC all-ones for WMMA

  const int m  = lane & 15;       // A row within strip / B,C column within tile
  const int hi = lane >> 4;       // half-wave selector
  const float ef = (MODE == 0) ? (1.0f + epsPtr[0]) : 0.0f;
  const float4* r4 = (const float4*)(in + (size_t)(rowBase + m) * GIN_H);
  const float4* g4 = (MODE == 0) ? (const float4*)(agg + (size_t)(rowBase + m) * GIN_H) : nullptr;

  // A fragments (ISA 16-bit A 16x32 layout). Per lane the swizzle is 4 runs of 8
  // consecutive k's starting at k0 = g*16 + hi*8 -> two float4 loads per run.
  v16h a0, a1;
#pragma unroll
  for (int g = 0; g < 4; ++g) {
    const int i0 = g * 4 + hi * 2;
    float4 p0 = r4[i0], p1 = r4[i0 + 1];
    float f[8];
    if (MODE == 0) {
      float4 q0 = g4[i0], q1 = g4[i0 + 1];
      f[0] = fmaf(ef, p0.x, q0.x); f[1] = fmaf(ef, p0.y, q0.y);
      f[2] = fmaf(ef, p0.z, q0.z); f[3] = fmaf(ef, p0.w, q0.w);
      f[4] = fmaf(ef, p1.x, q1.x); f[5] = fmaf(ef, p1.y, q1.y);
      f[6] = fmaf(ef, p1.z, q1.z); f[7] = fmaf(ef, p1.w, q1.w);
    } else {
      const float4* sc4 = (const float4*)sSc;
      const float4* sh4 = (const float4*)sSh;
      float4 s0 = sc4[i0], s1 = sc4[i0 + 1], t0 = sh4[i0], t1 = sh4[i0 + 1];
      f[0] = fmaxf(fmaf(p0.x, s0.x, t0.x), 0.0f); f[1] = fmaxf(fmaf(p0.y, s0.y, t0.y), 0.0f);
      f[2] = fmaxf(fmaf(p0.z, s0.z, t0.z), 0.0f); f[3] = fmaxf(fmaf(p0.w, s0.w, t0.w), 0.0f);
      f[4] = fmaxf(fmaf(p1.x, s1.x, t1.x), 0.0f); f[5] = fmaxf(fmaf(p1.y, s1.y, t1.y), 0.0f);
      f[6] = fmaxf(fmaf(p1.z, s1.z, t1.z), 0.0f); f[7] = fmaxf(fmaf(p1.w, s1.w, t1.w), 0.0f);
    }
    const int jb = (g & 1) * 8;
    if (g < 2) {
#pragma unroll
      for (int t = 0; t < 8; ++t) a0[jb + t] = (_Float16)f[t];
    } else {
#pragma unroll
      for (int t = 0; t < 8; ++t) a1[jb + t] = (_Float16)f[t];
    }
  }

#pragma unroll
  for (int nt = 0; nt < 4; ++nt) {
    const int col = nt * 16 + m;
    // B fragments (ISA 16-bit B 32x16 layout): per lane k = hi*8 + j (j<8), 16 + hi*8 + (j-8) (j>=8) (+32 for b1),
    // two runs of 8 halves of the transposed LDS row -> 2x ds_load_b128 each.
    const _Float16* wp = &Wt[col * WT_PITCH + hi * 8];
    v8h q0 = *(const v8h*)(wp);
    v8h q1 = *(const v8h*)(wp + 16);
    v8h q2 = *(const v8h*)(wp + 32);
    v8h q3 = *(const v8h*)(wp + 48);
    v16h b0, b1;
#pragma unroll
    for (int t = 0; t < 8; ++t) {
      b0[t] = q0[t]; b0[t + 8] = q1[t];
      b1[t] = q2[t]; b1[t + 8] = q3[t];
    }
    v8f c;
    const float cb = bias[col];                  // bias depends only on column
#pragma unroll
    for (int r = 0; r < 8; ++r) c[r] = cb;
    c = __builtin_amdgcn_wmma_f32_16x16x32_f16(false, a0, false, b0, (short)0, c, false, false);
    c = __builtin_amdgcn_wmma_f32_16x16x32_f16(false, a1, false, b1, (short)0, c, false, false);

    float s = 0.0f, s2 = 0.0f;
#pragma unroll
    for (int r = 0; r < 8; ++r) {
      int orow = rowBase + hi * 8 + r;           // ISA f32 C/D 16x16 layout
      out[(size_t)orow * GIN_H + col] = c[r];
      s  += c[r];
      s2 += c[r] * c[r];
    }
    atomicAdd(&colsum[col], s);
    atomicAdd(&colsq[col],  s2);
  }
}

// ------------------------------------------------- BN stats -> scale/shift
__global__ __launch_bounds__(64) void gin_bn_finalize(
    const float* __restrict__ sum, const float* __restrict__ sq,
    const float* __restrict__ gamma, const float* __restrict__ beta,
    float* __restrict__ scale, float* __restrict__ shift, float invn) {
  int c = threadIdx.x;
  float mmean = sum[c] * invn;
  float var   = sq[c] * invn - mmean * mmean;
  float s     = gamma[c] * rsqrtf(var + 1e-5f);
  scale[c] = s;
  shift[c] = beta[c] - mmean * s;
}

// ------------------------------------------------- fused BN2+ReLU + graph pooling
__global__ __launch_bounds__(256) void gin_bn_relu_pool(
    const float* __restrict__ u, const float* __restrict__ scale, const float* __restrict__ shift,
    const int* __restrict__ batch, float* __restrict__ hout, float* __restrict__ pooled) {
  unsigned tid = blockIdx.x * 256u + threadIdx.x;
  if (tid >= (unsigned)GIN_N * 16u) return;
  unsigned i  = tid >> 4;
  unsigned c4 = (tid & 15u) << 2;
  int g = batch[i];
  const float4 uv = *(const float4*)(u + (size_t)i * GIN_H + c4);
  float4 r;
  r.x = fmaxf(uv.x * scale[c4 + 0] + shift[c4 + 0], 0.0f);
  r.y = fmaxf(uv.y * scale[c4 + 1] + shift[c4 + 1], 0.0f);
  r.z = fmaxf(uv.z * scale[c4 + 2] + shift[c4 + 2], 0.0f);
  r.w = fmaxf(uv.w * scale[c4 + 3] + shift[c4 + 3], 0.0f);
  *(float4*)(hout + (size_t)i * GIN_H + c4) = r;
  float* pp = pooled + (size_t)g * GIN_H + c4;
  atomicAdd(pp + 0, r.x);
  atomicAdd(pp + 1, r.y);
  atomicAdd(pp + 2, r.z);
  atomicAdd(pp + 3, r.w);
}

// ------------------------------------------------- pool raw input features (layer-0 rep)
__global__ __launch_bounds__(256) void gin_pool(
    const float* __restrict__ x, const int* __restrict__ batch, float* __restrict__ pooled) {
  unsigned tid = blockIdx.x * 256u + threadIdx.x;
  if (tid >= (unsigned)GIN_N * 16u) return;
  unsigned i  = tid >> 4;
  unsigned c4 = (tid & 15u) << 2;
  int g = batch[i];
  const float4 xv = *(const float4*)(x + (size_t)i * GIN_H + c4);
  float* pp = pooled + (size_t)g * GIN_H + c4;
  atomicAdd(pp + 0, xv.x);
  atomicAdd(pp + 1, xv.y);
  atomicAdd(pp + 2, xv.z);
  atomicAdd(pp + 3, xv.w);
}

// ------------------------------------------------- per-layer heads
__global__ __launch_bounds__(32) void gin_readout(
    const float* __restrict__ pooled, const float* __restrict__ predW,
    const float* __restrict__ predb, float* __restrict__ logits) {
  int g = blockIdx.x;
  int c = threadIdx.x;
  if (c >= GIN_C) return;
  float acc = 0.0f;
  for (int l = 0; l <= GIN_L; ++l) {
    acc += predb[l * GIN_C + c];
    const float* pg = pooled + ((size_t)l * GIN_G + g) * GIN_H;
    const float* wl = predW + (size_t)l * GIN_H * GIN_C;
#pragma unroll 8
    for (int hh = 0; hh < GIN_H; ++hh) acc += pg[hh] * wl[hh * GIN_C + c];
  }
  logits[g * GIN_C + c] = acc;
}

// ================================================================ launcher
extern "C" void kernel_launch(void* const* d_in, const int* in_sizes, int n_in,
                              void* d_out, int out_size, void* d_ws, size_t ws_size,
                              hipStream_t stream) {
  const float* x     = (const float*)d_in[0];
  const int*   ei    = (const int*)d_in[1];
  const int*   src   = ei;
  const int*   dst   = ei + GIN_E;
  const int*   batch = (const int*)d_in[2];
  const float* ew    = (const float*)d_in[3];
  const float* eps   = (const float*)d_in[5];
  const float* W1    = (const float*)d_in[6];
  const float* b1    = (const float*)d_in[7];
  const float* W2    = (const float*)d_in[8];
  const float* b2    = (const float*)d_in[9];
  const float* bn1g  = (const float*)d_in[10];
  const float* bn1b  = (const float*)d_in[11];
  const float* bn2g  = (const float*)d_in[12];
  const float* bn2b  = (const float*)d_in[13];
  const float* predW = (const float*)d_in[14];
  const float* predb = (const float*)d_in[15];
  float* logits = (float*)d_out;

  float* ws = (float*)d_ws;
  const size_t NH = (size_t)GIN_N * GIN_H;
  float* agg    = ws;              // N*64; reused as 'u' (MLP-2 output) each layer
  float* t      = ws + NH;         // N*64  (MLP-1 output)
  float* hb0    = ws + 2 * NH;     // ping
  float* hb1    = ws + 3 * NH;     // pong
  float* pooled = ws + 4 * NH;                                  // (L+1)*G*64
  float* stats  = pooled + (size_t)(GIN_L + 1) * GIN_G * GIN_H; // 256 f32
  float* sum1 = stats,       *sq1 = stats + 64;
  float* sum2 = stats + 128, *sq2 = stats + 192;
  float* bns  = stats + 256;                                    // 256 f32
  float* sc1 = bns,       *sh1 = bns + 64;
  float* sc2 = bns + 128, *sh2 = bns + 192;

  const int poolN = (GIN_L + 1) * GIN_G * GIN_H;
  gin_zero<<<(poolN + 255) / 256, 256, 0, stream>>>(pooled, poolN);
  gin_pool<<<(GIN_N * 16) / 256, 256, 0, stream>>>(x, batch, pooled);

  const float* hin = x;
  float* houts[GIN_L] = {hb0, hb1, hb0};
  const int gemmBlocks = (GIN_N + 127) / 128;   // 8 waves x 16 rows per block
  const float invn = 1.0f / (float)GIN_N;

  for (int l = 0; l < GIN_L; ++l) {
    gin_zero<<<((int)NH + 255) / 256, 256, 0, stream>>>(agg, (int)NH);
    gin_zero<<<1, 256, 0, stream>>>(stats, 256);
    gin_aggregate<<<(GIN_E * 16) / 256, 256, 0, stream>>>(hin, src, dst, ew, agg);
    gin_gemm<0><<<gemmBlocks, 256, 0, stream>>>(
        hin, agg, eps + l, nullptr, nullptr,
        W1 + (size_t)l * GIN_H * GIN_H, b1 + l * GIN_H, t, sum1, sq1);
    gin_bn_finalize<<<1, GIN_H, 0, stream>>>(sum1, sq1, bn1g + l * GIN_H, bn1b + l * GIN_H,
                                             sc1, sh1, invn);
    gin_gemm<1><<<gemmBlocks, 256, 0, stream>>>(
        t, nullptr, nullptr, sc1, sh1,
        W2 + (size_t)l * GIN_H * GIN_H, b2 + l * GIN_H, agg /*u*/, sum2, sq2);
    gin_bn_finalize<<<1, GIN_H, 0, stream>>>(sum2, sq2, bn2g + l * GIN_H, bn2b + l * GIN_H,
                                             sc2, sh2, invn);
    gin_bn_relu_pool<<<(GIN_N * 16) / 256, 256, 0, stream>>>(
        agg, sc2, sh2, batch, houts[l], pooled + (size_t)(l + 1) * GIN_G * GIN_H);
    hin = houts[l];
  }

  gin_readout<<<GIN_G, 32, 0, stream>>>(pooled, predW, predb, logits);
}
